// RowAttentionWithPairBias_16716012716470
// MI455X (gfx1250) — matrix operand order repaired and checked by reading, not compiled, not run on hardware
//
#include <hip/hip_runtime.h>
#include <hip/hip_bf16.h>
#include <math.h>

// ---------------------------------------------------------------------------
// RowAttentionWithPairBias for MI455X (gfx1250, wave32, WMMA).
// R=1024, C_IN=256, H=8, D=32, C_Z=128.
// Memory-bound on streaming z (512 MB -> ~22us @ 23.3 TB/s); all GEMMs via
// v_wmma_f32_16x16x32_bf16; attention k/v tiles staged with
// global_load_async_to_lds_b128 (ASYNCcnt) -- CDNA5 async-to-LDS path.
// ---------------------------------------------------------------------------

typedef __attribute__((ext_vector_type(16))) __bf16         v16bf;
typedef __attribute__((ext_vector_type(16))) unsigned short v16us;
typedef __attribute__((ext_vector_type(8)))  float          v8f;

using u16 = unsigned short;

union frag16 { v16us u; v16bf b; };

__device__ inline u16 f32_to_bf16(float f) {
  unsigned u = __float_as_uint(f);
  unsigned r = (u + 0x7FFFu + ((u >> 16) & 1u)) >> 16;   // round-nearest-even
  return (u16)r;
}
__device__ inline float bf16f(u16 x) { return __uint_as_float(((unsigned)x) << 16); }

__device__ inline v8f wmma_bf16(const frag16& a, const frag16& b, v8f c) {
  // args: (neg_a, A, neg_b, B, c_mod, C, reuse_a, reuse_b)
  return __builtin_amdgcn_wmma_f32_16x16x32_bf16(false, a.b, false, b.b,
                                                 (short)0, c, false, false);
}

// reductions within a 16-lane group (rows 0-7 live in lanes 0-15, 8-15 in 16-31)
__device__ inline float rmax16(float v) {
  #pragma unroll
  for (int m = 1; m < 16; m <<= 1) v = fmaxf(v, __shfl_xor(v, m, 32));
  return v;
}
__device__ inline float rsum16(float v) {
  #pragma unroll
  for (int m = 1; m < 16; m <<= 1) v += __shfl_xor(v, m, 32);
  return v;
}
__device__ inline float rsum32(float v) {
  #pragma unroll
  for (int m = 1; m < 32; m <<= 1) v += __shfl_xor(v, m, 32);
  return v;
}

// ---------------------------------------------------------------------------
// 0) f32 -> bf16 weight conversion
// ---------------------------------------------------------------------------
__global__ void cvt_bf16_kernel(const float* __restrict__ src, u16* __restrict__ dst, int n) {
  int i = blockIdx.x * blockDim.x + threadIdx.x;
  if (i < n) dst[i] = f32_to_bf16(src[i]);
}

// ---------------------------------------------------------------------------
// 1) LayerNorm over C_IN=256, one block (256 thr) per row; bf16 output
// ---------------------------------------------------------------------------
__global__ void ln_m_kernel(const float* __restrict__ m, const float* __restrict__ w,
                            const float* __restrict__ b, u16* __restrict__ mn) {
  __shared__ float red[256];
  int row = blockIdx.x, t = threadIdx.x;
  float x = m[row * 256 + t];
  red[t] = x; __syncthreads();
  #pragma unroll
  for (int s = 128; s > 0; s >>= 1) { if (t < s) red[t] += red[t + s]; __syncthreads(); }
  float mu = red[0] * (1.f / 256.f);
  __syncthreads();
  float d = x - mu;
  red[t] = d * d; __syncthreads();
  #pragma unroll
  for (int s = 128; s > 0; s >>= 1) { if (t < s) red[t] += red[t + s]; __syncthreads(); }
  float var = red[0] * (1.f / 256.f);
  float y = d * rsqrtf(var + 1e-5f) * w[t] + b[t];
  mn[row * 256 + t] = f32_to_bf16(y);
}

// ---------------------------------------------------------------------------
// 2) Projection GEMM: [1024,256] x [256,256], one wave per 16x16 C tile.
//    mode 0: q (scale 1/sqrt(32), bf16 out)  1: k  2: v (bf16 out)
//    mode 3: g (sigmoid, f32 out)
// A layout (16-bit A 16x32): lane&15 = M row; VGPR i pair holds
//   K = ((i<4)? 2i : 16+2(i-4)) + 8*(lane>>4), K+1.
// B layout (16-bit B 32x16): lane&15 = N; element j = K, same K order as A.
// C layout: lane&15 = N; reg r = row r + 8*(lane>>4).
// ---------------------------------------------------------------------------
__global__ void proj_gemm_kernel(const u16* __restrict__ A, const u16* __restrict__ W,
                                 u16* __restrict__ outb, float* __restrict__ outf, int mode) {
  int lane = threadIdx.x;
  int gg = lane >> 4, row = lane & 15, nn = lane & 15;
  int m0 = blockIdx.x * 16, n0 = blockIdx.y * 16;
  v8f c = {};
  for (int k0 = 0; k0 < 256; k0 += 32) {
    frag16 a, b;
    const u16* ap = A + (m0 + row) * 256 + k0;
    #pragma unroll
    for (int i = 0; i < 8; ++i) {
      int kb = ((i < 4) ? 2 * i : 16 + 2 * (i - 4)) + 8 * gg;
      a.u[2 * i]     = ap[kb];
      a.u[2 * i + 1] = ap[kb + 1];
    }
    #pragma unroll
    for (int s = 0; s < 16; ++s)
      b.u[s] = W[(k0 + ((s < 8) ? s : 8 + s) + 8 * gg) * 256 + n0 + nn];
    c = wmma_bf16(a, b, c);
  }
  #pragma unroll
  for (int r = 0; r < 8; ++r) {
    int orow = m0 + r + 8 * gg, ocol = n0 + nn;
    float v = c[r];
    if (mode == 0) v *= 0.17677669529663687f;            // 1/sqrt(D)
    if (mode == 3) outf[orow * 256 + ocol] = 1.f / (1.f + __expf(-v));
    else           outb[orow * 256 + ocol] = f32_to_bf16(v);
  }
}

// ---------------------------------------------------------------------------
// 3) Pair bias: b[h][i][j] = sum_c LN(z[i,j,:])[c] * Wz[c][h], bf16 out.
//    One wave per (i,j): float4 per lane covers 128 channels (512 MB stream).
// ---------------------------------------------------------------------------
__global__ void pair_bias_kernel(const float* __restrict__ z,
                                 const float* __restrict__ lw, const float* __restrict__ lb,
                                 const float* __restrict__ Wz, u16* __restrict__ bias) {
  int lane = threadIdx.x & 31, wid = threadIdx.x >> 5;
  int i = blockIdx.x >> 7;
  int j = ((blockIdx.x & 127) << 3) | wid;
  const float* zp = z + ((size_t)i * 1024 + j) * 128;
  __builtin_prefetch(zp + 8 * 128, 0, 0);                // global_prefetch_b8 (next 8 pairs)
  float4 x = *(const float4*)(zp + lane * 4);
  float s  = x.x + x.y + x.z + x.w;
  float ss = x.x * x.x + x.y * x.y + x.z * x.z + x.w * x.w;
  s  = rsum32(s);
  ss = rsum32(ss);
  float mu  = s * (1.f / 128.f);
  float var = ss * (1.f / 128.f) - mu * mu;
  float rs  = rsqrtf(var + 1e-5f);
  float4 w4 = *(const float4*)(lw + lane * 4);
  float4 b4 = *(const float4*)(lb + lane * 4);
  float zn[4] = { (x.x - mu) * rs * w4.x + b4.x, (x.y - mu) * rs * w4.y + b4.y,
                  (x.z - mu) * rs * w4.z + b4.z, (x.w - mu) * rs * w4.w + b4.w };
  const float* wz = Wz + lane * 4 * 8;                    // Wz row-major [128][8]
  float acc[8] = {0, 0, 0, 0, 0, 0, 0, 0};
  #pragma unroll
  for (int e = 0; e < 4; ++e)
    #pragma unroll
    for (int h = 0; h < 8; ++h) acc[h] += zn[e] * wz[e * 8 + h];
  #pragma unroll
  for (int h = 0; h < 8; ++h) acc[h] = rsum32(acc[h]);
  float outv = acc[0];
  #pragma unroll
  for (int h = 1; h < 8; ++h) if (lane == h) outv = acc[h];
  if (lane < 8)
    bias[((size_t)lane << 20) + ((size_t)i << 10) + j] = f32_to_bf16(outv);
}

// ---------------------------------------------------------------------------
// 4) Split-K flash attention: one wave per (q-tile, head, key-split).
//    Each split covers 8 tiles of 32 keys (256 keys). k/v tiles staged to LDS
//    with global_load_async_to_lds_b128 (ASYNCcnt). Partials: unnormalized
//    numerator + per-row (m, l) for log-sum-exp combine.
// ---------------------------------------------------------------------------
__global__ void attn_split_kernel(const u16* __restrict__ q, const u16* __restrict__ k,
                                  const u16* __restrict__ v, const u16* __restrict__ bias,
                                  float* __restrict__ pnum, float* __restrict__ pm,
                                  float* __restrict__ pl) {
  __shared__ alignas(16) u16 qs[16 * 32];
  __shared__ alignas(16) u16 ks[32 * 32];
  __shared__ alignas(16) u16 vs[32 * 32];
  __shared__ alignas(16) u16 ps[16 * 32];
  int lane = threadIdx.x;
  int q0 = blockIdx.x * 16, h = blockIdx.y, sp = blockIdx.z;
  int gg = lane >> 4, row = lane & 15, nn = lane & 15;

  for (int t = lane; t < 512; t += 32)
    qs[t] = q[(q0 + (t >> 5)) * 256 + h * 32 + (t & 31)];
  frag16 aq;
  #pragma unroll
  for (int i = 0; i < 8; ++i) {
    int kb = ((i < 4) ? 2 * i : 16 + 2 * (i - 4)) + 8 * gg;
    aq.u[2 * i]     = qs[row * 32 + kb];
    aq.u[2 * i + 1] = qs[row * 32 + kb + 1];
  }

  float mrow[8], lrow[8];
  v8f acc0 = {}, acc1 = {};
  #pragma unroll
  for (int r = 0; r < 8; ++r) { mrow[r] = -INFINITY; lrow[r] = 0.f; }

  unsigned ldsk = (unsigned)(size_t)(void*)ks;            // LDS byte address (low 32b)
  unsigned ldsv = (unsigned)(size_t)(void*)vs;

  for (int kt = sp * 8; kt < sp * 8 + 8; ++kt) {
    int k0 = kt * 32;
    // Make sure prior-tile LDS reads have drained, then issue async copies:
    // 32 keys x 64B each = 128 x b128 chunks, 4 per lane.
    asm volatile("s_wait_dscnt 0x0" ::: "memory");
    #pragma unroll
    for (int it = 0; it < 4; ++it) {
      int idx = lane + 32 * it;                            // 0..127
      int key = idx >> 2, ch = idx & 3;
      unsigned goff = (unsigned)(((k0 + key) * 256 + h * 32) * 2 + ch * 16);
      unsigned lk = ldsk + (unsigned)(key * 64 + ch * 16);
      unsigned lv = ldsv + (unsigned)(key * 64 + ch * 16);
      asm volatile("global_load_async_to_lds_b128 %0, %1, %2"
                   :: "v"(lk), "v"(goff), "s"(k) : "memory");
      asm volatile("global_load_async_to_lds_b128 %0, %1, %2"
                   :: "v"(lv), "v"(goff), "s"(v) : "memory");
    }
    asm volatile("s_wait_asynccnt 0x0" ::: "memory");

    frag16 bk0, bk1;
    #pragma unroll
    for (int j = 0; j < 16; ++j) {                        // B: lane&15=N(key), elem=K(d)
      int kd = ((j < 8) ? j : 8 + j) + 8 * gg;
      bk0.u[j] = ks[nn * 32 + kd];
      bk1.u[j] = ks[(16 + nn) * 32 + kd];
    }
    v8f s0 = {}, s1 = {};
    s0 = wmma_bf16(aq, bk0, s0);
    s1 = wmma_bf16(aq, bk1, s1);

    const u16* bp = bias + ((size_t)h << 20) + k0;
    #pragma unroll
    for (int r = 0; r < 8; ++r) {
      int qr = q0 + r + 8 * gg;
      s0[r] += bf16f(bp[(size_t)qr * 1024 + nn]);
      s1[r] += bf16f(bp[(size_t)qr * 1024 + 16 + nn]);
    }

    float p0[8], p1[8];
    #pragma unroll
    for (int r = 0; r < 8; ++r) {
      float mloc = rmax16(fmaxf(s0[r], s1[r]));
      float mnew = fmaxf(mrow[r], mloc);
      float alpha = (mrow[r] == -INFINITY) ? 0.f : __expf(mrow[r] - mnew);
      p0[r] = __expf(s0[r] - mnew);
      p1[r] = __expf(s1[r] - mnew);
      lrow[r] = lrow[r] * alpha + rsum16(p0[r] + p1[r]);
      acc0[r] *= alpha;
      acc1[r] *= alpha;
      mrow[r] = mnew;
    }
    #pragma unroll
    for (int r = 0; r < 8; ++r) {                         // C layout -> LDS (row major)
      ps[(r + 8 * gg) * 32 + nn]      = f32_to_bf16(p0[r]);
      ps[(r + 8 * gg) * 32 + 16 + nn] = f32_to_bf16(p1[r]);
    }
    frag16 apf, bv0, bv1;
    #pragma unroll
    for (int i = 0; i < 8; ++i) {                         // A layout gather (transpose)
      int kb = ((i < 4) ? 2 * i : 16 + 2 * (i - 4)) + 8 * gg;
      apf.u[2 * i]     = ps[row * 32 + kb];
      apf.u[2 * i + 1] = ps[row * 32 + kb + 1];
    }
    #pragma unroll
    for (int j = 0; j < 16; ++j) {                        // B: lane&15=N(d), elem=K(key)
      int kk = ((j < 8) ? j : 8 + j) + 8 * gg;
      bv0.u[j] = vs[kk * 32 + nn];
      bv1.u[j] = vs[kk * 32 + 16 + nn];
    }
    acc0 = wmma_bf16(apf, bv0, acc0);
    acc1 = wmma_bf16(apf, bv1, acc1);
  }

  // write partials
  #pragma unroll
  for (int r = 0; r < 8; ++r) {
    int qr = q0 + r + 8 * gg;
    int c0 = qr * 256 + h * 32 + nn;
    pnum[((size_t)sp << 18) + c0]      = acc0[r];
    pnum[((size_t)sp << 18) + c0 + 16] = acc1[r];
  }
  if (nn == 0) {
    #pragma unroll
    for (int r = 0; r < 8; ++r) {
      int qr = q0 + r + 8 * gg;
      pm[((sp * 8 + h) << 10) + qr] = mrow[r];
      pl[((sp * 8 + h) << 10) + qr] = lrow[r];
    }
  }
}

// ---------------------------------------------------------------------------
// 4b) Combine split-K partials (log-sum-exp merge) + gating, bf16 out.
// ---------------------------------------------------------------------------
__global__ void attn_combine_kernel(const float* __restrict__ pnum, const float* __restrict__ pm,
                                    const float* __restrict__ pl, const float* __restrict__ gate,
                                    u16* __restrict__ o) {
  int idx = blockIdx.x * 256 + threadIdx.x;               // 0..262143
  int qr = idx >> 8, h = (idx & 255) >> 5;
  float mm[4], M = -INFINITY;
  #pragma unroll
  for (int s = 0; s < 4; ++s) {
    mm[s] = pm[((s * 8 + h) << 10) + qr];
    M = fmaxf(M, mm[s]);
  }
  float num = 0.f, den = 0.f;
  #pragma unroll
  for (int s = 0; s < 4; ++s) {
    float w = __expf(mm[s] - M);
    num += w * pnum[((size_t)s << 18) + idx];
    den += w * pl[((s * 8 + h) << 10) + qr];
  }
  o[idx] = f32_to_bf16(num / den * gate[idx]);
}

// ---------------------------------------------------------------------------
// 5) Output GEMM: out = o @ Wo + bo (f32 out)
// ---------------------------------------------------------------------------
__global__ void out_gemm_kernel(const u16* __restrict__ A, const u16* __restrict__ W,
                                const float* __restrict__ bo, float* __restrict__ out) {
  int lane = threadIdx.x;
  int gg = lane >> 4, row = lane & 15, nn = lane & 15;
  int m0 = blockIdx.x * 16, n0 = blockIdx.y * 16;
  v8f c = {};
  for (int k0 = 0; k0 < 256; k0 += 32) {
    frag16 a, b;
    const u16* ap = A + (m0 + row) * 256 + k0;
    #pragma unroll
    for (int i = 0; i < 8; ++i) {
      int kb = ((i < 4) ? 2 * i : 16 + 2 * (i - 4)) + 8 * gg;
      a.u[2 * i]     = ap[kb];
      a.u[2 * i + 1] = ap[kb + 1];
    }
    #pragma unroll
    for (int s = 0; s < 16; ++s)
      b.u[s] = W[(k0 + ((s < 8) ? s : 8 + s) + 8 * gg) * 256 + n0 + nn];
    c = wmma_bf16(a, b, c);
  }
  float bias = bo[n0 + nn];
  #pragma unroll
  for (int r = 0; r < 8; ++r)
    out[(m0 + r + 8 * gg) * 256 + n0 + nn] = c[r] + bias;
}

// ---------------------------------------------------------------------------
// Workspace layout (bytes)
// ---------------------------------------------------------------------------
static constexpr size_t OFF_MN = 0;                          // 1024*256 bf16
static constexpr size_t OFF_WQ = OFF_MN + 262144 * 2;        // 256*256 bf16 each
static constexpr size_t OFF_WK = OFF_WQ + 65536 * 2;
static constexpr size_t OFF_WV = OFF_WK + 65536 * 2;
static constexpr size_t OFF_WG = OFF_WV + 65536 * 2;
static constexpr size_t OFF_WO = OFF_WG + 65536 * 2;
static constexpr size_t OFF_Q  = OFF_WO + 65536 * 2;         // 1024*256 bf16
static constexpr size_t OFF_K  = OFF_Q + 262144 * 2;
static constexpr size_t OFF_V  = OFF_K + 262144 * 2;
static constexpr size_t OFF_G  = OFF_V + 262144 * 2;         // 1024*256 f32
static constexpr size_t OFF_O  = OFF_G + 262144 * 4;         // 1024*256 bf16
static constexpr size_t OFF_B  = OFF_O + 262144 * 2;         // 8*1024*1024 bf16 (16 MB)
static constexpr size_t OFF_PN = OFF_B + 8ull * 1024 * 1024 * 2;  // 4 x 1024*256 f32 (4 MB)
static constexpr size_t OFF_PM = OFF_PN + 4ull * 262144 * 4;      // 4*8*1024 f32
static constexpr size_t OFF_PL = OFF_PM + 32768 * 4;              // 4*8*1024 f32

extern "C" void kernel_launch(void* const* d_in, const int* in_sizes, int n_in,
                              void* d_out, int out_size, void* d_ws, size_t ws_size,
                              hipStream_t stream) {
  const float* m    = (const float*)d_in[0];
  const float* z    = (const float*)d_in[1];
  const float* lnmw = (const float*)d_in[2];
  const float* lnmb = (const float*)d_in[3];
  const float* lnzw = (const float*)d_in[4];
  const float* lnzb = (const float*)d_in[5];
  const float* Wz   = (const float*)d_in[6];
  const float* Wq   = (const float*)d_in[7];
  const float* Wk   = (const float*)d_in[8];
  const float* Wv   = (const float*)d_in[9];
  const float* Wg   = (const float*)d_in[10];
  // d_in[11] = bg: unused by the reference forward expression
  const float* Wo   = (const float*)d_in[12];
  const float* bo   = (const float*)d_in[13];
  float* out = (float*)d_out;

  char* ws = (char*)d_ws;
  u16*   mn_b  = (u16*)(ws + OFF_MN);
  u16*   wq_b  = (u16*)(ws + OFF_WQ);
  u16*   wk_b  = (u16*)(ws + OFF_WK);
  u16*   wv_b  = (u16*)(ws + OFF_WV);
  u16*   wg_b  = (u16*)(ws + OFF_WG);
  u16*   wo_b  = (u16*)(ws + OFF_WO);
  u16*   q_b   = (u16*)(ws + OFF_Q);
  u16*   k_b   = (u16*)(ws + OFF_K);
  u16*   v_b   = (u16*)(ws + OFF_V);
  float* g_f   = (float*)(ws + OFF_G);
  u16*   o_b   = (u16*)(ws + OFF_O);
  u16*   bias_b= (u16*)(ws + OFF_B);
  float* pn_f  = (float*)(ws + OFF_PN);
  float* pm_f  = (float*)(ws + OFF_PM);
  float* pl_f  = (float*)(ws + OFF_PL);

  // weight conversions
  cvt_bf16_kernel<<<256, 256, 0, stream>>>(Wq, wq_b, 65536);
  cvt_bf16_kernel<<<256, 256, 0, stream>>>(Wk, wk_b, 65536);
  cvt_bf16_kernel<<<256, 256, 0, stream>>>(Wv, wv_b, 65536);
  cvt_bf16_kernel<<<256, 256, 0, stream>>>(Wg, wg_b, 65536);
  cvt_bf16_kernel<<<256, 256, 0, stream>>>(Wo, wo_b, 65536);

  // LayerNorm(m)
  ln_m_kernel<<<1024, 256, 0, stream>>>(m, lnmw, lnmb, mn_b);

  // q/k/v/g projections (WMMA)
  proj_gemm_kernel<<<dim3(64, 16), 32, 0, stream>>>(mn_b, wq_b, q_b, nullptr, 0);
  proj_gemm_kernel<<<dim3(64, 16), 32, 0, stream>>>(mn_b, wk_b, k_b, nullptr, 1);
  proj_gemm_kernel<<<dim3(64, 16), 32, 0, stream>>>(mn_b, wv_b, v_b, nullptr, 2);
  proj_gemm_kernel<<<dim3(64, 16), 32, 0, stream>>>(mn_b, wg_b, nullptr, g_f, 3);

  // pair bias: streams all 512 MB of z once
  pair_bias_kernel<<<1024 * 128, 256, 0, stream>>>(z, lnzw, lnzb, Wz, bias_b);

  // split-K flash attention (async-to-LDS staging) + combine with gating
  attn_split_kernel<<<dim3(64, 8, 4), 32, 0, stream>>>(q_b, k_b, v_b, bias_b,
                                                       pn_f, pm_f, pl_f);
  attn_combine_kernel<<<1024, 256, 0, stream>>>(pn_f, pm_f, pl_f, g_f, o_b);

  // output projection
  out_gemm_kernel<<<dim3(64, 16), 32, 0, stream>>>(o_b, wo_b, bo, out);
}
